// CausalAttention1D_72447508349429
// MI455X (gfx1250) — compile-verified, matrix operand order checked by reading, not run on hardware
//
#include <hip/hip_runtime.h>
#include <stdint.h>

#define BDIM 4
#define CDIM 512
#define TDIM 4096

typedef __attribute__((ext_vector_type(16))) _Float16 v16h;
typedef __attribute__((ext_vector_type(8)))  float    v8f;

union Frag16 { v16h h; uint32_t u[8]; };
union U2H4  { uint2 u2; _Float16 h[4]; };
union H2U   { _Float16 h[2]; uint32_t u; };

// Load one 16x32 f16 A/B WMMA fragment row for this lane.
// rowPtr = start of this lane's matrix row (row = M for A, N for B).
// ISA layout: VGPR j holds K pair kk = 2*(j&3) + 8*hi + 16*(j>>2), hi = lane[4].
__device__ __forceinline__ v16h ldFragRow(const _Float16* rowPtr, unsigned hi) {
  Frag16 f;
  const uint32_t* p = reinterpret_cast<const uint32_t*>(rowPtr);
#pragma unroll
  for (int j = 0; j < 8; ++j) {
    unsigned kh = (unsigned)(j & 3) + hi * 4u + (unsigned)(j >> 2) * 8u; // kk/2
    f.u[j] = p[kh];
  }
  return f.h;
}

// ---------------------------------------------------------------------------
// Kernel A: fused Q/K/V projection.  out[b][t][o] = sum_c W[o][c]*x[b][c][t] + bias[o]
// Grid: (B*T/128, C/64, 3)  Block: 256 (8 waves; wave w = rows 16w..16w+15)
// ---------------------------------------------------------------------------
__global__ __launch_bounds__(256) void ca1d_qkv_proj(
    const float* __restrict__ x,
    const float* __restrict__ Wq, const float* __restrict__ bq,
    const float* __restrict__ Wk, const float* __restrict__ bk,
    const float* __restrict__ Wv, const float* __restrict__ bv,
    _Float16* __restrict__ qo, _Float16* __restrict__ ko, _Float16* __restrict__ vo)
{
  constexpr int BM = 128, BN = 64, BK = 32, LDK = 40; // LDK padded, even
  __shared__ _Float16 xs[BM * LDK];
  __shared__ _Float16 ws[BN * LDK];

  const int tid  = threadIdx.x;
  const int lane = tid & 31;
  const int wv   = tid >> 5;
  const unsigned hi = (lane >> 4) & 1u;
  const int nloc = lane & 15;

  const int bt0 = blockIdx.x * BM;
  const int b   = bt0 / TDIM;
  const int t0  = bt0 % TDIM;
  const int o0  = blockIdx.y * BN;

  const float* W; const float* bias; _Float16* out;
  if (blockIdx.z == 0)      { W = Wq; bias = bq; out = qo; }
  else if (blockIdx.z == 1) { W = Wk; bias = bk; out = ko; }
  else                      { W = Wv; bias = bv; out = vo; }

  v8f acc[4];
#pragma unroll
  for (int i = 0; i < 4; ++i) acc[i] = v8f{};

  for (int c0 = 0; c0 < CDIM; c0 += BK) {
    // xs[tt][kk] = x[b][c0+kk][t0+tt]; pack 2 k-adjacent f16 per b32 DS store
#pragma unroll
    for (int i = 0; i < (BM * BK) / (2 * 256); ++i) {
      int idx = tid + i * 256;
      int kk2 = (idx / BM) * 2;
      int tt  = idx % BM;
      const float* src = &x[((size_t)b * CDIM + (c0 + kk2)) * TDIM + (t0 + tt)];
      H2U p; p.h[0] = (_Float16)src[0]; p.h[1] = (_Float16)src[TDIM];
      *(uint32_t*)&xs[tt * LDK + kk2] = p.u;
    }
    // ws[oo][kk] = W[o0+oo][c0+kk]; pack 2 c-adjacent f16 per b32 DS store
#pragma unroll
    for (int i = 0; i < (BN * BK) / (2 * 256); ++i) {
      int idx = tid + i * 256;
      int oo = idx / (BK / 2);
      int kh = idx % (BK / 2);
      const float* src = &W[(size_t)(o0 + oo) * CDIM + c0 + kh * 2];
      H2U p; p.h[0] = (_Float16)src[0]; p.h[1] = (_Float16)src[1];
      *(uint32_t*)&ws[oo * LDK + kh * 2] = p.u;
    }
    __syncthreads();

    v16h a = ldFragRow(&xs[(wv * 16 + nloc) * LDK], hi);
#pragma unroll
    for (int nt = 0; nt < 4; ++nt) {
      v16h bfr = ldFragRow(&ws[(nt * 16 + nloc) * LDK], hi);
      acc[nt] = __builtin_amdgcn_wmma_f32_16x16x32_f16(
          false, a, false, bfr, (short)0, acc[nt], false, false);
    }
    __syncthreads();
  }

  // Epilogue: D-frag VGPR r -> row r (lanes 0-15) / r+8 (lanes 16-31), col = lane&15
#pragma unroll
  for (int nt = 0; nt < 4; ++nt) {
    float bvalue = bias[o0 + nt * 16 + nloc];
#pragma unroll
    for (int r = 0; r < 8; ++r) {
      int trow = t0 + wv * 16 + r + (int)hi * 8;
      out[((size_t)b * TDIM + trow) * CDIM + (o0 + nt * 16 + nloc)] =
          (_Float16)(acc[nt][r] + bvalue);
    }
  }
}

// ---------------------------------------------------------------------------
// Kernel B: causal flash attention.  h[b][t][c] = softmax(q k^T / sqrt(C)) v
// Grid: (T/32, B)  Block: 256 (8 waves).  Q frags live in registers.
// K tile staged with GLOBAL_LOAD_ASYNC_TO_LDS (ASYNCcnt path); V transposed
// by VALU into LDS concurrently with the in-flight async copy.
// ---------------------------------------------------------------------------
__global__ __launch_bounds__(256) void ca1d_flash_attn(
    const _Float16* __restrict__ q, const _Float16* __restrict__ k,
    const _Float16* __restrict__ v, _Float16* __restrict__ h)
{
  constexpr int BM = 32, BN = 64;
  constexpr float SCALE = 0.04419417382415922f; // 1/sqrt(512)
  constexpr float NEG_BIG = -3.0e38f;

  __shared__ _Float16 Ks[BN * CDIM];   // 64 KB  [key][channel]
  __shared__ _Float16 Vt[CDIM * BN];   // 64 KB  [channel][key]
  __shared__ float    Ss[BM * BN];     // 8 KB
  __shared__ _Float16 Ps[BM * BN];     // 4 KB
  __shared__ float rowM[BM], rowL[BM], rowC[BM];

  const int tid  = threadIdx.x;
  const int lane = tid & 31;
  const int wv   = tid >> 5;
  const unsigned hi = (lane >> 4) & 1u;
  const int nloc = lane & 15;

  const int t0 = blockIdx.x * BM;
  const int b  = blockIdx.y;

  const int mi = wv >> 2;         // which 16-query subtile this wave owns
  const int ni = wv & 3;          // 16-key N-tile for S; also channel group
  const int cb = ni * 128;        // channel base for P*V accumulation

  // Q fragments in registers: rows (t0 + 16*mi + nloc), all K=512 (16 k-steps)
  v16h qfr[16];
  {
    const _Float16* qrow =
        q + ((size_t)b * TDIM + t0 + mi * 16 + nloc) * CDIM;
#pragma unroll
    for (int kc = 0; kc < CDIM / 32; ++kc)
      qfr[kc] = ldFragRow(qrow + kc * 32, hi);
  }

  if (tid < BM) { rowM[tid] = NEG_BIG; rowL[tid] = 0.f; }

  v8f O[8];
#pragma unroll
  for (int i = 0; i < 8; ++i) O[i] = v8f{};

  __syncthreads();

  const int nkb = (t0 + BM - 1) / BN + 1; // causal: keys up to t0+BM-1
  for (int kb = 0; kb < nkb; ++kb) {
    const int s0 = kb * BN;

    // Stage K tile: flat 64 KB global->LDS via async copy (no VGPR bounce).
    {
      uint64_t gbase = (uint64_t)(uintptr_t)(k + ((size_t)b * TDIM + s0) * CDIM);
      unsigned lbase = (unsigned)(uintptr_t)&Ks[0];
#pragma unroll
      for (int i = 0; i < (BN * CDIM * 2) / (16 * 256); ++i) { // 16 x 16B/thread
        unsigned off   = (unsigned)(tid + i * 256) * 16u;
        unsigned laddr = lbase + off;
        uint64_t gaddr = gbase + off;
        asm volatile("global_load_async_to_lds_b128 %0, %1, off"
                     :: "v"(laddr), "v"(gaddr) : "memory");
      }
    }
    // Stage V tile transposed (VALU path, overlaps async copy): Vt[c][ss]
    {
      const uint2* src = (const uint2*)(v + ((size_t)b * TDIM + s0) * CDIM);
#pragma unroll
      for (int i = 0; i < (BN * CDIM / 4) / 256; ++i) {
        int idx = tid + i * 256;
        int ss = idx >> 7;            // / (CDIM/4)
        int c4 = (idx & 127) << 2;
        U2H4 d; d.u2 = src[idx];
#pragma unroll
        for (int e = 0; e < 4; ++e) Vt[(c4 + e) * BN + ss] = d.h[e];
      }
    }
    if (kb + 1 < nkb) { // speculative prefetch of next K/V block
      __builtin_prefetch(k + ((size_t)b * TDIM + s0 + BN) * CDIM, 0, 1);
      __builtin_prefetch(v + ((size_t)b * TDIM + s0 + BN) * CDIM, 0, 1);
    }
    asm volatile("s_wait_asynccnt 0x0" ::: "memory"); // K tile landed in LDS
    __syncthreads();

    // S tile (16x16 per wave): Q[mi] * K^T[ni], K = 512
    {
      v8f sacc = v8f{};
#pragma unroll 4
      for (int kc = 0; kc < CDIM / 32; ++kc) {
        v16h bb = ldFragRow(&Ks[(ni * 16 + nloc) * CDIM + kc * 32], hi);
        sacc = __builtin_amdgcn_wmma_f32_16x16x32_f16(
            false, qfr[kc], false, bb, (short)0, sacc, false, false);
      }
#pragma unroll
      for (int r = 0; r < 8; ++r) {
        int mloc = mi * 16 + r + (int)hi * 8;
        int nl   = ni * 16 + nloc;
        float val = sacc[r] * SCALE;
        if (s0 + nl > t0 + mloc) val = NEG_BIG; // causal mask
        Ss[mloc * BN + nl] = val;
      }
    }
    __syncthreads();

    // Online softmax: one thread per query row
    if (tid < BM) {
      const int r = tid;
      float mOld = rowM[r];
      float mNew = mOld;
#pragma unroll 8
      for (int j = 0; j < BN; ++j) mNew = fmaxf(mNew, Ss[r * BN + j]);
      float corr = __expf(mOld - mNew);
      float sum = 0.f;
#pragma unroll 8
      for (int j = 0; j < BN; ++j) {
        float p = __expf(Ss[r * BN + j] - mNew);
        Ps[r * BN + j] = (_Float16)p;
        sum += p;
      }
      rowL[r] = rowL[r] * corr + sum;
      rowM[r] = mNew;
      rowC[r] = corr;
    }
    __syncthreads();

    // Rescale running O, then O += P * V  (K = 64, two k-steps)
    float cf[8];
#pragma unroll
    for (int r = 0; r < 8; ++r) cf[r] = rowC[mi * 16 + r + (int)hi * 8];
#pragma unroll
    for (int nt = 0; nt < 8; ++nt)
#pragma unroll
      for (int r = 0; r < 8; ++r) O[nt][r] *= cf[r];

#pragma unroll
    for (int kc = 0; kc < BN / 32; ++kc) {
      v16h a = ldFragRow(&Ps[(mi * 16 + nloc) * BN + kc * 32], hi);
#pragma unroll
      for (int nt = 0; nt < 8; ++nt) {
        v16h bb = ldFragRow(&Vt[(cb + nt * 16 + nloc) * BN + kc * 32], hi);
        O[nt] = __builtin_amdgcn_wmma_f32_16x16x32_f16(
            false, a, false, bb, (short)0, O[nt], false, false);
      }
    }
    __syncthreads();
  }

  // Normalize and store h[b][t][c] (f16)
  float invl[8];
#pragma unroll
  for (int r = 0; r < 8; ++r) invl[r] = 1.0f / rowL[mi * 16 + r + (int)hi * 8];
#pragma unroll
  for (int nt = 0; nt < 8; ++nt) {
#pragma unroll
    for (int r = 0; r < 8; ++r) {
      int trow = t0 + mi * 16 + r + (int)hi * 8;
      h[((size_t)b * TDIM + trow) * CDIM + cb + nt * 16 + nloc] =
          (_Float16)(O[nt][r] * invl[r]);
    }
  }
}

// ---------------------------------------------------------------------------
// Kernel C: output projection + residual.
// out[b][o][t] = x[b][o][t] + bp[o] + sum_c Wp[o][c] * h[b][t][c]
// Grid: (B*T/128, C/64)  Block: 256
// ---------------------------------------------------------------------------
__global__ __launch_bounds__(256) void ca1d_out_proj(
    const _Float16* __restrict__ hin, const float* __restrict__ Wp,
    const float* __restrict__ bp, const float* __restrict__ x,
    float* __restrict__ out)
{
  constexpr int BM = 128, BN = 64, BK = 32, LDK = 40;
  __shared__ _Float16 hs[BM * LDK];
  __shared__ _Float16 ws[BN * LDK];

  const int tid  = threadIdx.x;
  const int lane = tid & 31;
  const int wv   = tid >> 5;
  const unsigned hi = (lane >> 4) & 1u;
  const int nloc = lane & 15;

  const int bt0 = blockIdx.x * BM;
  const int b   = bt0 / TDIM;
  const int t0  = bt0 % TDIM;
  const int o0  = blockIdx.y * BN;

  v8f acc[4];
#pragma unroll
  for (int i = 0; i < 4; ++i) acc[i] = v8f{};

  for (int c0 = 0; c0 < CDIM; c0 += BK) {
    // hs: h already f16 -> move 2 elements per b32 load/store, no converts
#pragma unroll
    for (int i = 0; i < (BM * BK) / (2 * 256); ++i) {
      int idx = tid + i * 256;
      int tt = idx / (BK / 2);
      int kh = idx % (BK / 2);
      uint32_t u = *(const uint32_t*)
          &hin[((size_t)b * TDIM + t0 + tt) * CDIM + c0 + kh * 2];
      *(uint32_t*)&hs[tt * LDK + kh * 2] = u;
    }
#pragma unroll
    for (int i = 0; i < (BN * BK) / (2 * 256); ++i) {
      int idx = tid + i * 256;
      int oo = idx / (BK / 2);
      int kh = idx % (BK / 2);
      const float* src = &Wp[(size_t)(o0 + oo) * CDIM + c0 + kh * 2];
      H2U p; p.h[0] = (_Float16)src[0]; p.h[1] = (_Float16)src[1];
      *(uint32_t*)&ws[oo * LDK + kh * 2] = p.u;
    }
    __syncthreads();

    v16h a = ldFragRow(&hs[(wv * 16 + nloc) * LDK], hi);
#pragma unroll
    for (int nt = 0; nt < 4; ++nt) {
      v16h bfr = ldFragRow(&ws[(nt * 16 + nloc) * LDK], hi);
      acc[nt] = __builtin_amdgcn_wmma_f32_16x16x32_f16(
          false, a, false, bfr, (short)0, acc[nt], false, false);
    }
    __syncthreads();
  }

#pragma unroll
  for (int nt = 0; nt < 4; ++nt) {
    int o = o0 + nt * 16 + nloc;
    float bvalue = bp[o];
#pragma unroll
    for (int r = 0; r < 8; ++r) {
      int trow = t0 + wv * 16 + r + (int)hi * 8;
      size_t idx = ((size_t)b * CDIM + o) * TDIM + trow;
      out[idx] = x[idx] + bvalue + acc[nt][r];
    }
  }
}

// ---------------------------------------------------------------------------
extern "C" void kernel_launch(void* const* d_in, const int* in_sizes, int n_in,
                              void* d_out, int out_size, void* d_ws, size_t ws_size,
                              hipStream_t stream) {
  (void)in_sizes; (void)n_in; (void)out_size; (void)ws_size;

  const float* x  = (const float*)d_in[0];
  const float* Wq = (const float*)d_in[1];
  const float* bq = (const float*)d_in[2];
  const float* Wk = (const float*)d_in[3];
  const float* bk = (const float*)d_in[4];
  const float* Wv = (const float*)d_in[5];
  const float* bv = (const float*)d_in[6];
  const float* Wp = (const float*)d_in[7];
  const float* bp = (const float*)d_in[8];
  float* out = (float*)d_out;

  const size_t nelem = (size_t)BDIM * TDIM * CDIM;
  _Float16* qb = (_Float16*)d_ws;     // f16 q [B][T][C]
  _Float16* kb = qb + nelem;          // f16 k [B][T][C]
  _Float16* vb = kb + nelem;          // f16 v [B][T][C]
  _Float16* hb = vb + nelem;          // f16 h [B][T][C]

  dim3 gA((BDIM * TDIM) / 128, CDIM / 64, 3);
  ca1d_qkv_proj<<<gA, 256, 0, stream>>>(x, Wq, bq, Wk, bk, Wv, bv, qb, kb, vb);

  dim3 gB(TDIM / 32, BDIM, 1);
  ca1d_flash_attn<<<gB, 256, 0, stream>>>(qb, kb, vb, hb);

  dim3 gC((BDIM * TDIM) / 128, CDIM / 64, 1);
  ca1d_out_proj<<<gC, 256, 0, stream>>>(hb, Wp, bp, x, out);
}
